// Build_str_70841190580558
// MI455X (gfx1250) — matrix operand order repaired and checked by reading, not compiled, not run on hardware
//
#include <hip/hip_runtime.h>
#include <hip/hip_bf16.h>

// ---------------------------------------------------------------------------
// Types for CDNA5 WMMA (wave32)
// ---------------------------------------------------------------------------
typedef __attribute__((ext_vector_type(16))) _Float16 v16h;
typedef __attribute__((ext_vector_type(8)))  _Float16 v8h;
typedef __attribute__((ext_vector_type(8)))  float    v8f;

#define TOKENS 65536      // B*N
#define CS 384
#define CH 128
#define NB 2
#define NA 7

__device__ __forceinline__ v16h cat16(v8h lo, v8h hi) {
    return __builtin_shufflevector(lo, hi, 0,1,2,3,4,5,6,7,8,9,10,11,12,13,14,15);
}

__device__ __forceinline__ v8f wmma_f16(v16h a, v16h b, v8f c) {
    // D = A(16x32 f16) * B(32x16 f16) + C(16x16 f32)
    return __builtin_amdgcn_wmma_f32_16x16x32_f16(false, a, false, b, (short)0, c,
                                                  false, false);
}

__device__ __forceinline__ v16h pack_a_relu(float4 a0, float4 a1, float4 b0, float4 b1) {
    v16h a;
    a[0]  = (_Float16)fmaxf(a0.x, 0.f); a[1]  = (_Float16)fmaxf(a0.y, 0.f);
    a[2]  = (_Float16)fmaxf(a0.z, 0.f); a[3]  = (_Float16)fmaxf(a0.w, 0.f);
    a[4]  = (_Float16)fmaxf(a1.x, 0.f); a[5]  = (_Float16)fmaxf(a1.y, 0.f);
    a[6]  = (_Float16)fmaxf(a1.z, 0.f); a[7]  = (_Float16)fmaxf(a1.w, 0.f);
    a[8]  = (_Float16)fmaxf(b0.x, 0.f); a[9]  = (_Float16)fmaxf(b0.y, 0.f);
    a[10] = (_Float16)fmaxf(b0.z, 0.f); a[11] = (_Float16)fmaxf(b0.w, 0.f);
    a[12] = (_Float16)fmaxf(b1.x, 0.f); a[13] = (_Float16)fmaxf(b1.y, 0.f);
    a[14] = (_Float16)fmaxf(b1.z, 0.f); a[15] = (_Float16)fmaxf(b1.w, 0.f);
    return a;
}

// A fragment (16x32) for this wave: row = own M row (m = lane&15),
// lanes 0-15: halfs = K[k0..k0+7], K[k0+16..k0+23]
// lanes 16-31: halfs = K[k0+8..k0+15], K[k0+24..k0+31]
__device__ __forceinline__ v16h load_a_f32_relu(const float* __restrict__ row,
                                                int k0, int hs) {
    const float4* q0 = (const float4*)(row + k0 + hs * 8);
    const float4* q1 = (const float4*)(row + k0 + 16 + hs * 8);
    return pack_a_relu(q0[0], q0[1], q1[0], q1[1]);
}

// Same, but also writes the raw f32 data through to `orow` (fused passthrough
// of `s` to the output buffer -- saves a whole 100MB HBM re-read vs memcpy).
__device__ __forceinline__ v16h load_a_f32_relu_store(const float* __restrict__ row,
                                                      float* __restrict__ orow,
                                                      int k0, int hs) {
    const float4* q0 = (const float4*)(row + k0 + hs * 8);
    const float4* q1 = (const float4*)(row + k0 + 16 + hs * 8);
    float4 a0 = q0[0], a1 = q0[1], b0 = q1[0], b1 = q1[1];
    float4* o0 = (float4*)(orow + k0 + hs * 8);
    float4* o1 = (float4*)(orow + k0 + 16 + hs * 8);
    o0[0] = a0; o0[1] = a1; o1[0] = b0; o1[1] = b1;
    return pack_a_relu(a0, a1, b0, b1);
}

// A fragment out of LDS-staged f16 activations (already ReLU'd).
__device__ __forceinline__ v16h load_a_lds(const _Float16* row, int k0, int hs) {
    v8h lo = *(const v8h*)(row + k0 + hs * 8);
    v8h hi = *(const v8h*)(row + k0 + 16 + hs * 8);
    return cat16(lo, hi);
}

// B fragment (32x16): WT is weights transposed [N][K] in f16.
// lane's column n = nTile*16 + (lane&15); K = k0 + (lane>>4)*8 .. +7, then k0 + 16 + (lane>>4)*8 .. +7.
__device__ __forceinline__ v16h load_b_t(const _Float16* __restrict__ WT, int ldk,
                                         int nTile, int k0, int lane) {
    int n = nTile * 16 + (lane & 15);
    const _Float16* p = WT + (size_t)n * ldk + k0 + (lane >> 4) * 8;
    v8h lo = *(const v8h*)p;
    v8h hi = *(const v8h*)(p + 16);
    return cat16(lo, hi);
}

// ---------------------------------------------------------------------------
// Weight transpose + f32->f16 convert:  dst[n*K+k] = (n<N) ? src[k*N+n] : 0
// ---------------------------------------------------------------------------
__global__ void transpose_f16_kernel(const float* __restrict__ src,
                                     _Float16* __restrict__ dst,
                                     int K, int N, int Npad) {
    int idx = blockIdx.x * blockDim.x + threadIdx.x;
    if (idx >= Npad * K) return;
    int n = idx / K;
    int k = idx - n * K;
    dst[idx] = (n < N) ? (_Float16)src[k * N + n] : (_Float16)0.f;
}

// ---------------------------------------------------------------------------
// Fused angle-resnet: 5-layer GEMM chain, x kept in registers per wave.
// Block = 8 waves = 128 token rows; each wave owns 16 rows x 128 cols.
// Wave-private LDS rows -> no barriers needed anywhere.
// ---------------------------------------------------------------------------
__global__ __launch_bounds__(256) void resnet_wmma_kernel(
    const float* __restrict__ s, const float* __restrict__ s_init,
    const _Float16* __restrict__ Wf16,
    const float* __restrict__ b_init, const float* __restrict__ b_in,
    const float* __restrict__ b1, const float* __restrict__ b2,
    const float* __restrict__ b_out,
    float* __restrict__ out_unnorm, float* __restrict__ out_angles,
    float* __restrict__ out_s) {
    __shared__ _Float16 xr[128 * CH];   // 32 KB staging (rows wave-private)

    const int lane = threadIdx.x & 31;
    const int wave = threadIdx.x >> 5;
    const int m    = lane & 15;   // A-layout row / B,C-layout column
    const int hs   = lane >> 4;   // half-select
    const int rowBase = blockIdx.x * 128 + wave * 16;

    const _Float16* WinitT = Wf16;                 // [128][384]
    const _Float16* WinT   = Wf16 + 49152;         // [128][384]
    const _Float16* W1T    = Wf16 + 98304;         // [2][128][128]
    const _Float16* W2T    = Wf16 + 131072;        // [2][128][128]
    const _Float16* WoutT  = Wf16 + 163840;        // [16][128] (padded 14->16)

    v8f xacc[8];
#pragma unroll
    for (int nt = 0; nt < 8; ++nt)
#pragma unroll
        for (int g = 0; g < 8; ++g) xacc[nt][g] = 0.f;

    // ---- layer 1: x = relu(s_init)@Winit + relu(s)@Win + b_init + b_in ----
    {
        const float* arow = s_init + (size_t)(rowBase + m) * CS;
        for (int k0 = 0; k0 < CS; k0 += 32) {
            __builtin_prefetch(arow + k0 + 128, 0, 1);   // stream-ahead
            v16h a = load_a_f32_relu(arow, k0, hs);
#pragma unroll
            for (int nt = 0; nt < 8; ++nt) {
                v16h b = load_b_t(WinitT, CS, nt, k0, lane);
                xacc[nt] = wmma_f16(a, b, xacc[nt]);
            }
        }
        arow = s + (size_t)(rowBase + m) * CS;
        float* orow = out_s + (size_t)(rowBase + m) * CS;
        for (int k0 = 0; k0 < CS; k0 += 32) {
            __builtin_prefetch(arow + k0 + 128, 0, 1);
            v16h a = load_a_f32_relu_store(arow, orow, k0, hs);
#pragma unroll
            for (int nt = 0; nt < 8; ++nt) {
                v16h b = load_b_t(WinT, CS, nt, k0, lane);
                xacc[nt] = wmma_f16(a, b, xacc[nt]);
            }
        }
#pragma unroll
        for (int nt = 0; nt < 8; ++nt) {
            float bv = b_init[nt * 16 + m] + b_in[nt * 16 + m];
#pragma unroll
            for (int g = 0; g < 8; ++g) xacc[nt][g] += bv;
        }
    }

    // stage relu(x) -> f16 LDS (C layout -> row-major)
#pragma unroll
    for (int nt = 0; nt < 8; ++nt) {
        int col = nt * 16 + m;
#pragma unroll
        for (int g = 0; g < 8; ++g) {
            int r = wave * 16 + g + hs * 8;
            xr[r * CH + col] = (_Float16)fmaxf(xacc[nt][g], 0.f);
        }
    }

    const _Float16* aro = xr + (size_t)(wave * 16 + m) * CH;

    // ---- resnet blocks ----
    for (int blk = 0; blk < NB; ++blk) {
        const _Float16* w1 = W1T + blk * (CH * CH);
        const _Float16* w2 = W2T + blk * (CH * CH);

        v8f h[8];
#pragma unroll
        for (int nt = 0; nt < 8; ++nt)
#pragma unroll
            for (int g = 0; g < 8; ++g) h[nt][g] = 0.f;

        for (int k0 = 0; k0 < CH; k0 += 32) {
            v16h a = load_a_lds(aro, k0, hs);
#pragma unroll
            for (int nt = 0; nt < 8; ++nt)
                h[nt] = wmma_f16(a, load_b_t(w1, CH, nt, k0, lane), h[nt]);
        }
        // h = relu(h + b1); stage over xr (old x no longer needed in LDS)
#pragma unroll
        for (int nt = 0; nt < 8; ++nt) {
            int col = nt * 16 + m;
            float bv = b1[blk * CH + col];
#pragma unroll
            for (int g = 0; g < 8; ++g) {
                int r = wave * 16 + g + hs * 8;
                xr[r * CH + col] = (_Float16)fmaxf(h[nt][g] + bv, 0.f);
            }
        }

        v8f a2[8];
#pragma unroll
        for (int nt = 0; nt < 8; ++nt)
#pragma unroll
            for (int g = 0; g < 8; ++g) a2[nt][g] = 0.f;

        for (int k0 = 0; k0 < CH; k0 += 32) {
            v16h a = load_a_lds(aro, k0, hs);
#pragma unroll
            for (int nt = 0; nt < 8; ++nt)
                a2[nt] = wmma_f16(a, load_b_t(w2, CH, nt, k0, lane), a2[nt]);
        }
        // x += a2 + b2 ; restage relu(x)
#pragma unroll
        for (int nt = 0; nt < 8; ++nt) {
            int col = nt * 16 + m;
            float bv = b2[blk * CH + col];
#pragma unroll
            for (int g = 0; g < 8; ++g) {
                xacc[nt][g] += a2[nt][g] + bv;
                int r = wave * 16 + g + hs * 8;
                xr[r * CH + col] = (_Float16)fmaxf(xacc[nt][g], 0.f);
            }
        }
    }

    // ---- output layer: relu(x) @ w_out + b_out, N=14 (padded to 16) ----
    v8f o;
#pragma unroll
    for (int g = 0; g < 8; ++g) o[g] = 0.f;
    for (int k0 = 0; k0 < CH; k0 += 32) {
        v16h a = load_a_lds(aro, k0, hs);
        o = wmma_f16(a, load_b_t(WoutT, CH, 0, k0, lane), o);
    }
    int col = m;                    // 0..15, valid < 14; col = angle*2 + comp
    float bo = (col < 14) ? b_out[col] : 0.f;
#pragma unroll
    for (int g = 0; g < 8; ++g) {
        float v = o[g] + bo;
        float p = __shfl_xor(v, 1, 32);     // partner component (sin<->cos)
        float nrm = sqrtf(fmaxf(v * v + p * p, 1e-8f));
        float an = v / nrm;
        if (col < 14) {
            int token = rowBase + g + hs * 8;
            out_unnorm[(size_t)token * 14 + col] = v;
            out_angles[(size_t)token * 14 + col] = an;
        }
    }
}

// ---------------------------------------------------------------------------
// Per-token frame composition, quaternion, side-chain frames, atom positions
// ---------------------------------------------------------------------------
__device__ __forceinline__ void mm3(const float* A, const float* B, float* C) {
#pragma unroll
    for (int i = 0; i < 3; ++i)
#pragma unroll
        for (int j = 0; j < 3; ++j)
            C[i * 3 + j] = A[i * 3 + 0] * B[0 * 3 + j] +
                           A[i * 3 + 1] * B[1 * 3 + j] +
                           A[i * 3 + 2] * B[2 * 3 + j];
}
__device__ __forceinline__ void mv3(const float* A, const float* v, float* o) {
#pragma unroll
    for (int i = 0; i < 3; ++i)
        o[i] = A[i * 3 + 0] * v[0] + A[i * 3 + 1] * v[1] + A[i * 3 + 2] * v[2];
}

__global__ __launch_bounds__(128) void frames_kernel(
    const float* __restrict__ rot, const float* __restrict__ trans,
    const int* __restrict__ aatype, const float* __restrict__ dframes,
    const int* __restrict__ gidx, const float* __restrict__ amask,
    const float* __restrict__ lit, const float* __restrict__ angles,
    float* __restrict__ frames7, float* __restrict__ sc44,
    float* __restrict__ pos) {
    int t = blockIdx.x * blockDim.x + threadIdx.x;
    if (t >= TOKENS) return;

    const float* Rp = rot + (size_t)t * 9;
    float Rb[9];
#pragma unroll
    for (int i = 0; i < 9; ++i) Rb[i] = Rp[i];
    float bt[3] = { trans[t * 3 + 0] * 10.f, trans[t * 3 + 1] * 10.f,
                    trans[t * 3 + 2] * 10.f };

    // quaternion (leading eigenvector of the K matrix == unit quaternion)
    {
        float r00 = Rb[0], r01 = Rb[1], r02 = Rb[2];
        float r10 = Rb[3], r11 = Rb[4], r12 = Rb[5];
        float r20 = Rb[6], r21 = Rb[7], r22 = Rb[8];
        float qw, qx, qy, qz;
        float tr = r00 + r11 + r22;
        if (tr > 0.f) {
            float S = sqrtf(tr + 1.f) * 2.f;
            qw = 0.25f * S; qx = (r21 - r12) / S;
            qy = (r02 - r20) / S; qz = (r10 - r01) / S;
        } else if (r00 > r11 && r00 > r22) {
            float S = sqrtf(1.f + r00 - r11 - r22) * 2.f;
            qw = (r21 - r12) / S; qx = 0.25f * S;
            qy = (r01 + r10) / S; qz = (r02 + r20) / S;
        } else if (r11 > r22) {
            float S = sqrtf(1.f + r11 - r00 - r22) * 2.f;
            qw = (r02 - r20) / S; qx = (r01 + r10) / S;
            qy = 0.25f * S; qz = (r12 + r21) / S;
        } else {
            float S = sqrtf(1.f + r22 - r00 - r11) * 2.f;
            qw = (r10 - r01) / S; qx = (r02 + r20) / S;
            qy = (r12 + r21) / S; qz = 0.25f * S;
        }
        float qn = rsqrtf(qw * qw + qx * qx + qy * qy + qz * qz);
        float* f7 = frames7 + (size_t)t * 7;
        f7[0] = qw * qn; f7[1] = qx * qn; f7[2] = qy * qn; f7[3] = qz * qn;
        f7[4] = bt[0]; f7[5] = bt[1]; f7[6] = bt[2];
    }

    int aat = aatype[t];
    const float* df  = dframes + (size_t)aat * 8 * 16;
    const float* ang = angles + (size_t)t * 14;

    float fR[8][9], ft[8][3];
#pragma unroll
    for (int f = 0; f < 8; ++f) {
        float sn = (f == 0) ? 0.f : ang[(f - 1) * 2 + 0];
        float cs = (f == 0) ? 1.f : ang[(f - 1) * 2 + 1];
        const float* d = df + f * 16;
#pragma unroll
        for (int i = 0; i < 3; ++i) {
            float d0 = d[i * 4 + 0], d1 = d[i * 4 + 1], d2 = d[i * 4 + 2];
            fR[f][i * 3 + 0] = d0;                   // rotx col0 = e0
            fR[f][i * 3 + 1] = cs * d1 + sn * d2;    // col1 = (0,c,s)
            fR[f][i * 3 + 2] = -sn * d1 + cs * d2;   // col2 = (0,-s,c)
            ft[f][i] = d[i * 4 + 3];
        }
    }
    // chain: f5 = f4∘f5 ; f6 = f5∘f6 ; f7 = f6∘f7
#pragma unroll
    for (int f = 5; f < 8; ++f) {
        float tmpR[9], tmpt[3];
        mm3(fR[f - 1], fR[f], tmpR);
        mv3(fR[f - 1], ft[f], tmpt);
#pragma unroll
        for (int i = 0; i < 9; ++i) fR[f][i] = tmpR[i];
#pragma unroll
        for (int i = 0; i < 3; ++i) ft[f][i] = tmpt[i] + ft[f - 1][i];
    }

    // global frames + sc44 output (all register-resident, constant indexing)
    float Rg[8][9], tg[8][3];
#pragma unroll
    for (int f = 0; f < 8; ++f) {
        mm3(Rb, fR[f], Rg[f]);
        float tv[3];
        mv3(Rb, ft[f], tv);
#pragma unroll
        for (int i = 0; i < 3; ++i) tg[f][i] = tv[i] + bt[i];
        float* o = sc44 + (size_t)t * 128 + f * 16;
#pragma unroll
        for (int i = 0; i < 3; ++i) {
#pragma unroll
            for (int j = 0; j < 3; ++j) o[i * 4 + j] = Rg[f][i * 3 + j];
            o[i * 4 + 3] = tg[f][i];
        }
        o[12] = 0.f; o[13] = 0.f; o[14] = 0.f; o[15] = 1.f;
    }

    // atom positions: select frame by group index via unrolled register
    // selects (avoids dynamic indexing -> avoids scratch spills of Rg/tg)
    const int* gi = gidx + (size_t)aat * 14;
    const float* lp = lit + (size_t)aat * 14 * 3;
    const float* mk = amask + (size_t)aat * 14;
    float* po = pos + (size_t)t * 42;
#pragma unroll 2
    for (int a = 0; a < 14; ++a) {
        int g = gi[a];
        float Ra[9], ta[3];
#pragma unroll
        for (int i = 0; i < 9; ++i) Ra[i] = Rg[0][i];
#pragma unroll
        for (int i = 0; i < 3; ++i) ta[i] = tg[0][i];
#pragma unroll
        for (int f = 1; f < 8; ++f) {
            bool hit = (g == f);
#pragma unroll
            for (int i = 0; i < 9; ++i) Ra[i] = hit ? Rg[f][i] : Ra[i];
#pragma unroll
            for (int i = 0; i < 3; ++i) ta[i] = hit ? tg[f][i] : ta[i];
        }
        float v[3] = { lp[a * 3 + 0], lp[a * 3 + 1], lp[a * 3 + 2] };
        float p[3];
        mv3(Ra, v, p);
        float msk = mk[a];
        po[a * 3 + 0] = (p[0] + ta[0]) * msk;
        po[a * 3 + 1] = (p[1] + ta[1]) * msk;
        po[a * 3 + 2] = (p[2] + ta[2]) * msk;
    }
}

// ---------------------------------------------------------------------------
// Launch
// ---------------------------------------------------------------------------
extern "C" void kernel_launch(void* const* d_in, const int* in_sizes, int n_in,
                              void* d_out, int out_size, void* d_ws, size_t ws_size,
                              hipStream_t stream) {
    (void)in_sizes; (void)n_in; (void)out_size; (void)ws_size;

    const float* s       = (const float*)d_in[0];
    const float* s_init  = (const float*)d_in[1];
    const float* rot     = (const float*)d_in[2];
    const float* trans   = (const float*)d_in[3];
    const int*   aatype  = (const int*)d_in[4];
    const float* dframes = (const float*)d_in[5];
    const int*   gidx    = (const int*)d_in[6];
    const float* amask   = (const float*)d_in[7];
    const float* lit     = (const float*)d_in[8];
    const float* w_init  = (const float*)d_in[9];
    const float* b_init  = (const float*)d_in[10];
    const float* w_in    = (const float*)d_in[11];
    const float* b_in    = (const float*)d_in[12];
    const float* blk_w1  = (const float*)d_in[13];
    const float* blk_b1  = (const float*)d_in[14];
    const float* blk_w2  = (const float*)d_in[15];
    const float* blk_b2  = (const float*)d_in[16];
    const float* w_out   = (const float*)d_in[17];
    const float* b_out   = (const float*)d_in[18];

    float* out = (float*)d_out;
    float* o_frames7 = out + 0;          //  65536*7
    float* o_sc44    = out + 458752;     //  65536*8*16
    float* o_unnorm  = out + 8847360;    //  65536*14
    float* o_angles  = out + 9764864;    //  65536*14
    float* o_pos     = out + 10682368;   //  65536*42
    float* o_s       = out + 13434880;   //  65536*384

    _Float16* W = (_Float16*)d_ws;       // transposed f16 weights (~324 KB)

    // weight transposes (f32 -> f16, [K][N] -> [Npad][K])
    auto launchT = [&](const float* src, _Float16* dst, int K, int N, int Npad) {
        int n = Npad * K;
        transpose_f16_kernel<<<(n + 255) / 256, 256, 0, stream>>>(src, dst, K, N, Npad);
    };
    launchT(w_init,          W + 0,              384, 128, 128);
    launchT(w_in,            W + 49152,          384, 128, 128);
    launchT(blk_w1,          W + 98304,          128, 128, 128);
    launchT(blk_w1 + 16384,  W + 98304 + 16384,  128, 128, 128);
    launchT(blk_w2,          W + 131072,         128, 128, 128);
    launchT(blk_w2 + 16384,  W + 131072 + 16384, 128, 128, 128);
    launchT(w_out,           W + 163840,         128, 14, 16);

    // fused WMMA resnet: 512 blocks x 8 waves, 128 token rows per block
    // (also streams s through to o_s, fused with the layer-1 loads)
    resnet_wmma_kernel<<<512, 256, 0, stream>>>(
        s, s_init, W, b_init, b_in, blk_b1, blk_b2, b_out,
        o_unnorm, o_angles, o_s);

    // per-token geometry
    frames_kernel<<<TOKENS / 128, 128, 0, stream>>>(
        rot, trans, aatype, dframes, gidx, amask, lit, o_angles,
        o_frames7, o_sc44, o_pos);
}
